// LNGRUCell_24515673325921
// MI455X (gfx1250) — compile-verified, matrix operand order checked by reading, not run on hardware
//
#include <hip/hip_runtime.h>
#include <hip/hip_bf16.h>

// ---------- types ----------
typedef __attribute__((ext_vector_type(8)))  __bf16 v8bf;
typedef __attribute__((ext_vector_type(16))) __bf16 v16bf;
typedef __attribute__((ext_vector_type(8)))  float  v8f;

#define HDIM 1024
#define BDIM 4096
#define N3   3072
#define LDSROW 40   // padded LDS row stride (elements) -> conflict-free b128 reads

// ---------- workspace layout (bytes) ----------
static constexpr size_t OFF_XBF   = 0;
static constexpr size_t OFF_HBF   = 8388608;
static constexpr size_t OFF_WT    = 16777216;
static constexpr size_t OFF_UT    = 23068672;
static constexpr size_t OFF_XW    = 29360128;
static constexpr size_t OFF_HU    = 79691776;
static constexpr size_t OFF_STATS = 130023424;

// ---------- CDNA5 async global->LDS helpers ----------
// GVS mode: mem = SGPR64 base + VGPR32 offset; VDST VGPR carries LDS byte addr.
// Tracked by ASYNCcnt (completes in issue order).
__device__ __forceinline__ unsigned lds_u32(const void* p) {
  // shared-aperture flat address: low 32 bits are the wave-relative LDS offset
  return (unsigned)(unsigned long long)p;
}
__device__ __forceinline__ void async_ld_b128(unsigned ldsaddr, const void* base, unsigned voff) {
  asm volatile("global_load_async_to_lds_b128 %0, %1, %2"
               :: "v"(ldsaddr), "v"(voff), "s"(base) : "memory");
}
__device__ __forceinline__ void wait_async_le4() {
  asm volatile("s_wait_asynccnt 4" ::: "memory");
}
__device__ __forceinline__ void wait_async_0() {
  asm volatile("s_wait_asynccnt 0" ::: "memory");
}

// ---------- kernel 1: f32 -> bf16 ----------
__global__ __launch_bounds__(256) void lngru_cvt_bf16(const float* __restrict__ src,
                                                      __bf16* __restrict__ dst, int n) {
  int i = blockIdx.x * blockDim.x + threadIdx.x;
  if (i < n) dst[i] = (__bf16)src[i];
}

// ---------- kernel 2: transpose-convert W(1024x3072 f32) -> Wt(3072x1024 bf16) ----------
__global__ __launch_bounds__(256) void lngru_transpose_bf16(const float* __restrict__ W,
                                                            __bf16* __restrict__ Wt) {
  __shared__ __bf16 t[32][33];
  const int bx = blockIdx.x * 32;  // n
  const int by = blockIdx.y * 32;  // k
  const int tx = threadIdx.x;
  const int ty = threadIdx.y;
  #pragma unroll
  for (int r = ty; r < 32; r += 8)
    t[r][tx] = (__bf16)W[(size_t)(by + r) * N3 + bx + tx];
  __syncthreads();
  #pragma unroll
  for (int r = ty; r < 32; r += 8)
    Wt[(size_t)(bx + r) * HDIM + by + tx] = t[tx][r];
}

// ---------- kernel 3: bf16 WMMA GEMM, C(f32) = A(MxK) * Bt(NxK)^T ----------
// Double-buffered LDS, async global->LDS prefetch of tile k+1 overlapped with
// WMMAs of tile k.
__global__ __launch_bounds__(256) void lngru_gemm_bf16(const __bf16* __restrict__ A,
                                                       const __bf16* __restrict__ Bt,
                                                       float* __restrict__ C,
                                                       int M, int N, int K) {
  __shared__ __bf16 sA[2][128 * LDSROW];
  __shared__ __bf16 sB[2][128 * LDSROW];

  const int tid  = threadIdx.x;
  const int lane = tid & 31;
  const int wave = tid >> 5;
  const int wm   = (wave & 3) * 32;   // wave M offset in block tile
  const int wn   = (wave >> 2) * 64;  // wave N offset in block tile
  const int l15  = lane & 15;
  const int hi   = lane >> 4;         // 0: K 0..7/16..23, 1: K 8..15/24..31
  const long rowM0 = (long)blockIdx.y * 128;
  const long rowN0 = (long)blockIdx.x * 128;

  // per-thread tile-load chunks: 2 for A, 2 for B, each 16B
  const int c0 = tid,       r0 = c0 >> 2, k0e = (c0 & 3) * 8;
  const int c1 = tid + 256, r1 = c1 >> 2, k1e = (c1 & 3) * 8;

  // global byte offsets relative to A / Bt base (add kk*64 per K-step)
  const unsigned aOff0 = (unsigned)(((rowM0 + r0) * K + k0e) * 2);
  const unsigned aOff1 = (unsigned)(((rowM0 + r1) * K + k1e) * 2);
  const unsigned bOff0 = (unsigned)(((rowN0 + r0) * K + k0e) * 2);
  const unsigned bOff1 = (unsigned)(((rowN0 + r1) * K + k1e) * 2);

  // LDS byte addresses for both buffers
  unsigned lA0[2], lA1[2], lB0[2], lB1[2];
  #pragma unroll
  for (int b = 0; b < 2; ++b) {
    lA0[b] = lds_u32(&sA[b][r0 * LDSROW + k0e]);
    lA1[b] = lds_u32(&sA[b][r1 * LDSROW + k1e]);
    lB0[b] = lds_u32(&sB[b][r0 * LDSROW + k0e]);
    lB1[b] = lds_u32(&sB[b][r1 * LDSROW + k1e]);
  }

  v8f acc[2][4];
  #pragma unroll
  for (int tm = 0; tm < 2; ++tm)
    #pragma unroll
    for (int tn = 0; tn < 4; ++tn)
      acc[tm][tn] = (v8f){0.f,0.f,0.f,0.f,0.f,0.f,0.f,0.f};

  const int NK = K >> 5;  // K-steps of 32

  // prologue: async-fetch tile 0 into buffer 0
  async_ld_b128(lA0[0], A,  aOff0);
  async_ld_b128(lA1[0], A,  aOff1);
  async_ld_b128(lB0[0], Bt, bOff0);
  async_ld_b128(lB1[0], Bt, bOff1);

  for (int k = 0; k < NK; ++k) {
    const int cur = k & 1;
    if (k + 1 < NK) {
      const unsigned kb = (unsigned)(k + 1) * 64u;  // bytes per K-step
      const int nxt = cur ^ 1;
      async_ld_b128(lA0[nxt], A,  aOff0 + kb);
      async_ld_b128(lA1[nxt], A,  aOff1 + kb);
      async_ld_b128(lB0[nxt], Bt, bOff0 + kb);
      async_ld_b128(lB1[nxt], Bt, bOff1 + kb);
      wait_async_le4();   // tile k landed; tile k+1 still in flight
    } else {
      wait_async_0();
    }
    __syncthreads();      // tile-k LDS writes visible to all waves

    // fragments per ISA 16-bit A layout:
    // lanes 0-15: K 0..7 then 16..23 ; lanes 16-31: K 8..15 then 24..31
    v16bf af[2], bfg[4];
    #pragma unroll
    for (int tm = 0; tm < 2; ++tm) {
      const int r = wm + tm * 16 + l15;
      v8bf lo = *(const v8bf*)&sA[cur][r * LDSROW + hi * 8];
      v8bf hh = *(const v8bf*)&sA[cur][r * LDSROW + hi * 8 + 16];
      af[tm] = __builtin_shufflevector(lo, hh, 0,1,2,3,4,5,6,7,8,9,10,11,12,13,14,15);
    }
    #pragma unroll
    for (int tn = 0; tn < 4; ++tn) {
      const int r = wn + tn * 16 + l15;
      v8bf lo = *(const v8bf*)&sB[cur][r * LDSROW + hi * 8];
      v8bf hh = *(const v8bf*)&sB[cur][r * LDSROW + hi * 8 + 16];
      bfg[tn] = __builtin_shufflevector(lo, hh, 0,1,2,3,4,5,6,7,8,9,10,11,12,13,14,15);
    }

    #pragma unroll
    for (int tm = 0; tm < 2; ++tm)
      #pragma unroll
      for (int tn = 0; tn < 4; ++tn)
        acc[tm][tn] = __builtin_amdgcn_wmma_f32_16x16x32_bf16(
            false, af[tm], false, bfg[tn], (short)0, acc[tm][tn], false, false);

    __syncthreads();      // all waves done reading buf[cur] before it is re-filled
  }

  // epilogue: C/D layout -> m = r + 8*(lane>>4), n = lane&15
  #pragma unroll
  for (int tm = 0; tm < 2; ++tm)
    #pragma unroll
    for (int tn = 0; tn < 4; ++tn)
      #pragma unroll
      for (int r = 0; r < 8; ++r) {
        const long row = rowM0 + wm + tm * 16 + r + hi * 8;
        const long col = rowN0 + wn + tn * 16 + l15;
        C[row * N + col] = acc[tm][tn][r];
      }
}

// ---------- kernel 4: per-row mean / rstd over 3072 cols ----------
__global__ __launch_bounds__(256) void lngru_row_stats(const float* __restrict__ XW,
                                                       const float* __restrict__ HU,
                                                       float* __restrict__ stats) {
  __shared__ float s1[256];
  __shared__ float s2[256];
  const int tid = threadIdx.x;
  const int row = blockIdx.x;
  const float* src = (blockIdx.y == 0 ? XW : HU) + (size_t)row * N3;
  float a = 0.f, b = 0.f;
  #pragma unroll
  for (int j = tid; j < N3; j += 256) {
    float v = src[j];
    a += v; b += v * v;
  }
  s1[tid] = a; s2[tid] = b;
  __syncthreads();
  for (int off = 128; off > 0; off >>= 1) {
    if (tid < off) { s1[tid] += s1[tid + off]; s2[tid] += s2[tid + off]; }
    __syncthreads();
  }
  if (tid == 0) {
    const float mu  = s1[0] * (1.f / N3);
    const float var = s2[0] * (1.f / N3) - mu * mu;
    const size_t base = (size_t)blockIdx.y * 2 * BDIM;
    stats[base + row]        = mu;
    stats[base + BDIM + row] = rsqrtf(var + 1e-5f);
  }
}

// ---------- kernel 5: fused LN + GRU gates ----------
__global__ __launch_bounds__(256) void lngru_gates(const float* __restrict__ XW,
                                                   const float* __restrict__ HU,
                                                   const float* __restrict__ stats,
                                                   const float* __restrict__ hx,
                                                   const float* __restrict__ b,
                                                   const float* __restrict__ Wg,
                                                   const float* __restrict__ Wb,
                                                   const float* __restrict__ Ug,
                                                   const float* __restrict__ Ub,
                                                   float* __restrict__ out) {
  const int idx = blockIdx.x * blockDim.x + threadIdx.x;
  if (idx >= BDIM * HDIM) return;
  const int i = idx >> 10;
  const int j = idx & (HDIM - 1);
  const float mu_x = stats[i];
  const float rs_x = stats[BDIM + i];
  const float mu_h = stats[2 * BDIM + i];
  const float rs_h = stats[3 * BDIM + i];
  const size_t rbase = (size_t)i * N3;

  float xw[3], hu[3];
  #pragma unroll
  for (int k = 0; k < 3; ++k) {
    const int col = k * HDIM + j;
    xw[k] = (XW[rbase + col] - mu_x) * rs_x * Wg[col] + Wb[col];
    hu[k] = (HU[rbase + col] - mu_h) * rs_h * Ug[col] + Ub[col];
  }
  const float z = 1.f / (1.f + __expf(-(xw[0] + hu[0] + b[j])));
  const float r = 1.f / (1.f + __expf(-(xw[1] + hu[1] + b[HDIM + j])));
  const float hc = tanhf(r * hu[2] + xw[2] + b[2 * HDIM + j]);
  out[idx] = (1.f - z) * hc + z * hx[idx];
}

// ---------- launcher ----------
extern "C" void kernel_launch(void* const* d_in, const int* in_sizes, int n_in,
                              void* d_out, int out_size, void* d_ws, size_t ws_size,
                              hipStream_t stream) {
  const float* input = (const float*)d_in[0];
  const float* hx    = (const float*)d_in[1];
  const float* W     = (const float*)d_in[2];
  const float* U     = (const float*)d_in[3];
  const float* b     = (const float*)d_in[4];
  const float* Wg    = (const float*)d_in[5];
  const float* Wb    = (const float*)d_in[6];
  const float* Ug    = (const float*)d_in[7];
  const float* Ub    = (const float*)d_in[8];
  float* out = (float*)d_out;

  char* ws = (char*)d_ws;
  __bf16* Xbf = (__bf16*)(ws + OFF_XBF);
  __bf16* Hbf = (__bf16*)(ws + OFF_HBF);
  __bf16* Wt  = (__bf16*)(ws + OFF_WT);
  __bf16* Ut  = (__bf16*)(ws + OFF_UT);
  float*  XW  = (float*)(ws + OFF_XW);
  float*  HU  = (float*)(ws + OFF_HU);
  float*  st  = (float*)(ws + OFF_STATS);

  const int nElem = BDIM * HDIM;

  lngru_cvt_bf16<<<(nElem + 255) / 256, 256, 0, stream>>>(input, Xbf, nElem);
  lngru_cvt_bf16<<<(nElem + 255) / 256, 256, 0, stream>>>(hx, Hbf, nElem);

  lngru_transpose_bf16<<<dim3(N3 / 32, HDIM / 32), dim3(32, 8), 0, stream>>>(W, Wt);
  lngru_transpose_bf16<<<dim3(N3 / 32, HDIM / 32), dim3(32, 8), 0, stream>>>(U, Ut);

  lngru_gemm_bf16<<<dim3(N3 / 128, BDIM / 128), 256, 0, stream>>>(Xbf, Wt, XW, BDIM, N3, HDIM);
  lngru_gemm_bf16<<<dim3(N3 / 128, BDIM / 128), 256, 0, stream>>>(Hbf, Ut, HU, BDIM, N3, HDIM);

  lngru_row_stats<<<dim3(BDIM, 2), 256, 0, stream>>>(XW, HU, st);

  lngru_gates<<<(nElem + 255) / 256, 256, 0, stream>>>(XW, HU, st, hx, b, Wg, Wb, Ug, Ub, out);
}
